// RCNN_83442624626796
// MI455X (gfx1250) — matrix operand order repaired and checked by reading, not compiled, not run on hardware
//
#include <hip/hip_runtime.h>
#include <hip/hip_bf16.h>

typedef __attribute__((ext_vector_type(16))) _Float16 v16h;
typedef __attribute__((ext_vector_type(8)))  _Float16 v8h;
typedef __attribute__((ext_vector_type(2)))  _Float16 v2h;
typedef __attribute__((ext_vector_type(8)))  float    v8f;
typedef __attribute__((ext_vector_type(4)))  float    v4f;

union H16 { v16h v; v8h h[2]; };

#define BM 128
#define BN 128
#define BK 32
#define LDA 40   // padded LDS stride (halves) for A tile: conflict-free ds_load_b128
#define LDB 40   // padded LDS stride (halves) for B^T tile

// ---------------------------------------------------------------------------
// Kernel 1: SlowROIPool (floor/ceil crop + adaptive max pool 7x7), f32 -> f16
// Output X[roi][c*49 + oy*7 + ox]  (matches reshape(N, C*7*7))
// ---------------------------------------------------------------------------
__global__ __launch_bounds__(256) void roi_pool_f16(
    const float* __restrict__ feat, const int* __restrict__ rois,
    _Float16* __restrict__ X)
{
  const int roi = blockIdx.x;
  const int b  =  rois[roi * 5 + 0];
  const int x1 =  rois[roi * 5 + 1] >> 4;          // floor div 16 (non-negative)
  const int y1 =  rois[roi * 5 + 2] >> 4;
  const int x2 = (rois[roi * 5 + 3] + 15) >> 4;    // ceil div 16
  const int y2 = (rois[roi * 5 + 4] + 15) >> 4;
  const int h = y2 - y1;
  const int w = x2 - x1;

  int ys[7], ye[7], xs[7], xe[7];
  for (int o = 0; o < 7; ++o) {
    int a  = y1 + (o * h) / 7;
    int bb = y1 + ((o + 1) * h + 6) / 7;
    ys[o] = a  < 0 ? 0 : (a  > 14 ? 14 : a);
    ye[o] = bb < 0 ? 0 : (bb > 14 ? 14 : bb);
    a  = x1 + (o * w) / 7;
    bb = x1 + ((o + 1) * w + 6) / 7;
    xs[o] = a  < 0 ? 0 : (a  > 14 ? 14 : a);
    xe[o] = bb < 0 ? 0 : (bb > 14 ? 14 : bb);
  }

  const float* fb = feat + (size_t)b * (512 * 196);
  for (int c = threadIdx.x; c < 512; c += 256) {
    const float* fc = fb + c * 196;
    _Float16* xo = X + (size_t)roi * 25088 + c * 49;
    for (int oy = 0; oy < 7; ++oy) {
      for (int ox = 0; ox < 7; ++ox) {
        float m = -3.4028234663852886e38f;
        for (int y = ys[oy]; y < ye[oy]; ++y)
          for (int x = xs[ox]; x < xe[ox]; ++x)
            m = fmaxf(m, fc[y * 14 + x]);
        xo[oy * 7 + ox] = (_Float16)m;
      }
    }
  }
}

// ---------------------------------------------------------------------------
// Kernel 2: generic tiled GEMM on v_wmma_f32_16x16x32_f16.
//   out = [relu](A(f16, MxK) * W(f32 -> f16 in flight, KxN) + bias)
//   128x128x32 tiles, 256 threads = 8 waves, each wave: 2x4 WMMA accumulators.
//   Requires K % 32 == 0 and M % 128 == 0 (true for all call sites).
//   Software-pipelined: next tile's global loads overlap the WMMA burst.
// ---------------------------------------------------------------------------
template <bool WF, bool WH, bool RELU>
__global__ __launch_bounds__(256) void gemm_f16_wmma(
    const _Float16* __restrict__ A, const float* __restrict__ W,
    const float* __restrict__ bias,
    float* __restrict__ outF, _Float16* __restrict__ outH,
    int M, int N, int K)
{
  __shared__ _Float16 As[BM * LDA];   // A tile, row-major [m][k]
  __shared__ _Float16 Bs[BN * LDB];   // B tile, transposed [n][k]

  const int tid  = threadIdx.x;
  const int lane = tid & 31;
  const int wave = tid >> 5;
  const int wm   = wave >> 1;     // 0..3 : 32-row band inside tile
  const int wn   = wave & 1;      // 0..1 : 64-col band inside tile
  const int lm   = lane & 15;
  const int half = lane >> 4;

  // blockIdx.x = M tile (fast-varying -> blocks sharing a weight slab co-run)
  const int m0 = blockIdx.x * BM;
  const int n0 = blockIdx.y * BN;
  const bool fulln = (n0 + BN) <= N;   // block-uniform

  // A tile loader mapping: 128 rows x 32 halves, 16 halves per thread
  const int a_row = tid >> 1;
  const int a_col = (tid & 1) << 4;
  // B tile loader mapping: 2 consecutive K rows x 8 N cols per thread
  const int b_k = (tid & 15) << 1;
  const int b_n = (tid >> 4) << 3;

  const v8f vzero = {0.f, 0.f, 0.f, 0.f, 0.f, 0.f, 0.f, 0.f};
  v8f acc[2][4];
#pragma unroll
  for (int i = 0; i < 2; ++i)
#pragma unroll
    for (int j = 0; j < 4; ++j) acc[i][j] = vzero;

  const v4f fzero = {0.f, 0.f, 0.f, 0.f};

  auto loadA = [&](int k0, v8h& a0, v8h& a1) {
    const _Float16* ag = A + (size_t)(m0 + a_row) * K + (k0 + a_col);
    a0 = *(const v8h*)ag;
    a1 = *(const v8h*)(ag + 8);
  };
  auto loadB = [&](int k0, v4f& l0, v4f& l1, v4f& h0, v4f& h1) {
    const float* wg0 = W + (size_t)(k0 + b_k) * N + (n0 + b_n);
    const float* wg1 = wg0 + N;
    if (fulln) {                       // uniform fast path: 4x b128 loads
      l0 = *(const v4f*)wg0; l1 = *(const v4f*)(wg0 + 4);
      h0 = *(const v4f*)wg1; h1 = *(const v4f*)(wg1 + 4);
    } else {                           // edge blocks (head GEMMs) only
      l0 = fzero; l1 = fzero; h0 = fzero; h1 = fzero;
#pragma unroll
      for (int j = 0; j < 4; ++j) {
        if (n0 + b_n + j < N)     { l0[j] = wg0[j];     h0[j] = wg1[j]; }
        if (n0 + b_n + 4 + j < N) { l1[j] = wg0[4 + j]; h1[j] = wg1[4 + j]; }
      }
    }
  };
  auto stage = [&](v8h a0, v8h a1, v4f l0, v4f l1, v4f h0, v4f h1) {
    *(v8h*)&As[a_row * LDA + a_col]     = a0;
    *(v8h*)&As[a_row * LDA + a_col + 8] = a1;
#pragma unroll
    for (int j = 0; j < 4; ++j) {
      v2h p = { (_Float16)l0[j], (_Float16)h0[j] };
      *(v2h*)&Bs[(b_n + j) * LDB + b_k] = p;
      v2h q = { (_Float16)l1[j], (_Float16)h1[j] };
      *(v2h*)&Bs[(b_n + 4 + j) * LDB + b_k] = q;
    }
  };

  v8h ca0, ca1;
  v4f cl0, cl1, ch0, ch1;
  loadA(0, ca0, ca1);
  loadB(0, cl0, cl1, ch0, ch1);

  for (int k0 = 0; k0 < K; k0 += BK) {
    stage(ca0, ca1, cl0, cl1, ch0, ch1);
    __syncthreads();

    // prefetch next tile into registers (overlaps with WMMA below)
    if (k0 + BK < K) {
      loadA(k0 + BK, ca0, ca1);
      loadB(k0 + BK, cl0, cl1, ch0, ch1);
    }

    // ---- build WMMA fragments per ISA VGPR layouts ----
    H16 af[2], bf[4];
#pragma unroll
    for (int i = 0; i < 2; ++i) {
      const int m = wm * 32 + i * 16 + lm;
      // A 16x32 f16: lanes 0-15 -> K chunks {0..7,16..23}; lanes 16-31 -> {8..15,24..31}
      af[i].h[0] = *(const v8h*)&As[m * LDA + half * 8];
      af[i].h[1] = *(const v8h*)&As[m * LDA + 16 + half * 8];
    }
#pragma unroll
    for (int j = 0; j < 4; ++j) {
      const int n = wn * 64 + j * 16 + lm;
      // B 32x16 f16: lanes 0-15 -> K chunks {0..7,16..23}; lanes 16-31 -> {8..15,24..31}
      bf[j].h[0] = *(const v8h*)&Bs[n * LDB + half * 8];
      bf[j].h[1] = *(const v8h*)&Bs[n * LDB + 16 + half * 8];
    }

#pragma unroll
    for (int i = 0; i < 2; ++i)
#pragma unroll
      for (int j = 0; j < 4; ++j)
        acc[i][j] = __builtin_amdgcn_wmma_f32_16x16x32_f16(
            false, af[i].v, false, bf[j].v, (short)0, acc[i][j], false, false);

    __syncthreads();
  }

  // ---- epilogue: bias + optional relu; C/D layout row = r + 8*half, col = lm ----
#pragma unroll
  for (int i = 0; i < 2; ++i) {
#pragma unroll
    for (int j = 0; j < 4; ++j) {
      const int row0 = m0 + wm * 32 + i * 16 + half * 8;
      const int col  = n0 + wn * 64 + j * 16 + lm;
      if (fulln || col < N) {
        const float bb = bias[col];
#pragma unroll
        for (int r = 0; r < 8; ++r) {
          float v = acc[i][j][r] + bb;
          if (RELU) v = fmaxf(v, 0.f);
          const size_t idx = (size_t)(row0 + r) * N + col;
          if (WF) outF[idx] = v;
          if (WH) outH[idx] = (_Float16)v;
        }
      }
    }
  }
}

// ---------------------------------------------------------------------------
extern "C" void kernel_launch(void* const* d_in, const int* in_sizes, int n_in,
                              void* d_out, int out_size, void* d_ws, size_t ws_size,
                              hipStream_t stream) {
  (void)in_sizes; (void)n_in; (void)out_size; (void)ws_size;
  const float* feat   = (const float*)d_in[0];
  const int*   rois   = (const int*)  d_in[1];
  const float* fc1_w  = (const float*)d_in[2];
  const float* fc1_b  = (const float*)d_in[3];
  const float* fc2_w  = (const float*)d_in[4];
  const float* fc2_b  = (const float*)d_in[5];
  const float* cls_w  = (const float*)d_in[6];
  const float* cls_b  = (const float*)d_in[7];
  const float* bbox_w = (const float*)d_in[8];
  const float* bbox_b = (const float*)d_in[9];
  float* out = (float*)d_out;

  char* ws = (char*)d_ws;
  _Float16* X  = (_Float16*)ws;                                   // 1024 x 25088 f16 (51,380,224 B)
  _Float16* H1 = (_Float16*)(ws + (size_t)51380224);              // 1024 x 4096  f16 ( 8,388,608 B)
  _Float16* H2 = (_Float16*)(ws + (size_t)51380224 + 8388608);    // 1024 x 4096  f16

  roi_pool_f16<<<1024, 256, 0, stream>>>(feat, rois, X);

  // FC1: 1024 x 4096 x 25088 (grid.x = M tiles so weight-slab sharers co-run)
  gemm_f16_wmma<false, true, true><<<dim3(8, 32), 256, 0, stream>>>(
      X,  fc1_w, fc1_b, nullptr, H1, 1024, 4096, 25088);
  // FC2: 1024 x 4096 x 4096
  gemm_f16_wmma<false, true, true><<<dim3(8, 32), 256, 0, stream>>>(
      H1, fc2_w, fc2_b, nullptr, H2, 1024, 4096, 4096);
  // cls head: 1024 x 21 x 4096 -> out[0 .. 21504)
  gemm_f16_wmma<true, false, false><<<dim3(8, 1), 256, 0, stream>>>(
      H2, cls_w,  cls_b,  out, nullptr, 1024, 21, 4096);
  // bbox head: 1024 x 84 x 4096 -> out[21504 .. 107520)
  gemm_f16_wmma<true, false, false><<<dim3(8, 1), 256, 0, stream>>>(
      H2, bbox_w, bbox_b, out + 21504, nullptr, 1024, 84, 4096);
}
